// EdgeAwareAE_48619029790898
// MI455X (gfx1250) — matrix operand order repaired and checked by reading, not compiled, not run on hardware
//
#include <hip/hip_runtime.h>
#include <hip/hip_bf16.h>
#include <cstdint>

#define NNODES   10000
#define NEDGES   80000
#define IN_DIM   128
#define EDGE_DIM 16
#define HID      256
#define OUTD     128
#define H1       4

typedef __attribute__((ext_vector_type(16))) _Float16 v16h;
typedef __attribute__((ext_vector_type(8)))  _Float16 v8h;
typedef __attribute__((ext_vector_type(4)))  _Float16 v4h;
typedef __attribute__((ext_vector_type(2)))  _Float16 v2h;
typedef __attribute__((ext_vector_type(8)))  float    v8f;

// ---- monotonic float <-> ordered-uint transform for atomic segment-max ----
__device__ __forceinline__ unsigned f2ord(float f){
  unsigned u = __float_as_uint(f);
  return (u & 0x80000000u) ? ~u : (u | 0x80000000u);
}
__device__ __forceinline__ float ord2f(unsigned u){
  return (u & 0x80000000u) ? __uint_as_float(u & 0x7FFFFFFFu) : __uint_as_float(~u);
}

// ---- zero fill (works for both float 0.0 and uint 0 init) ----
__global__ void zero_u32_kernel(unsigned* __restrict__ p, int n){
  int i = blockIdx.x * blockDim.x + threadIdx.x;
  if (i < n) p[i] = 0u;
}

__global__ void relu_kernel(float* __restrict__ p, int n){
  int i = blockIdx.x * blockDim.x + threadIdx.x;
  if (i < n) p[i] = fmaxf(p[i], 0.f);
}

// ---- WMMA GEMM: C[M,Nn] = A[M,K] @ B[K,Nn] + bias[Nn] ----
// 128 threads = 4 waves; block tile 64x64; wave w -> rows [16w,16w+16) x 64 cols.
// Requires: K % 32 == 0, Nn % 64 == 0. M guarded on store only (A rows clamped:
// garbage rows >= M only feed C rows >= M, which are never stored).
// Double-buffered LDS; fragments are 16B contiguous runs -> ds_load_b128.
__global__ __launch_bounds__(128)
void gemm_bias_wmma_kernel(const float* __restrict__ A, const float* __restrict__ B,
                           const float* __restrict__ bias, float* __restrict__ C,
                           int M, int K, int Nn){
  __shared__ _Float16 sA [2][64][40];   // [buf][row][k]
  __shared__ _Float16 sBt[2][64][40];   // [buf][col][k]  (B transposed)
  const int tid  = threadIdx.x;
  const int lane = tid & 31;
  const int wave = tid >> 5;
  const int row0 = blockIdx.y * 64;
  const int col0 = blockIdx.x * 64;

  const v8f vzero = {0.f,0.f,0.f,0.f,0.f,0.f,0.f,0.f};
  v8f acc[4];
#pragma unroll
  for (int t = 0; t < 4; ++t) acc[t] = vzero;

  // loop-invariant fragment coordinates
  const int am  = wave*16 + (lane & 15);     // A row this lane owns
  const int akb = (lane >> 4) * 8;           // A k-half base (0 or 8)
  const int bkb = (lane >> 4) * 8;           // B k-half base (0 or 8)

  // staging registers (tile in flight)
  float4 aR[4];
  float  bR[8][2];

  auto issue_loads = [&](int kb){
#pragma unroll
    for (int r = 0; r < 4; ++r){
      int idx = tid + 128*r;                 // 512 quads
      int ar = idx >> 3;                     // 0..63
      int aq = idx & 7;                      // quad within 32 k
      int row = row0 + ar;
      row = row < M ? row : M - 1;           // clamp: branch-free, always valid
      aR[r] = *(const float4*)&A[(size_t)row*K + kb + 4*aq];
    }
#pragma unroll
    for (int r = 0; r < 8; ++r){
      int idx = tid + 128*r;                 // 1024 (kpair, col) tasks
      int bc = idx & 63;                     // col (coalesced)
      int bp = idx >> 6;                     // kpair 0..15
      bR[r][0] = B[(size_t)(kb + 2*bp    )*Nn + col0 + bc];
      bR[r][1] = B[(size_t)(kb + 2*bp + 1)*Nn + col0 + bc];
    }
  };
  auto store_lds = [&](int buf){
#pragma unroll
    for (int r = 0; r < 4; ++r){
      int idx = tid + 128*r;
      int ar = idx >> 3, aq = idx & 7;
      v4h h4 = { (_Float16)aR[r].x, (_Float16)aR[r].y,
                 (_Float16)aR[r].z, (_Float16)aR[r].w };
      *(v4h*)&sA[buf][ar][4*aq] = h4;
    }
#pragma unroll
    for (int r = 0; r < 8; ++r){
      int idx = tid + 128*r;
      int bc = idx & 63, bp = idx >> 6;
      v2h p = { (_Float16)bR[r][0], (_Float16)bR[r][1] };
      *(v2h*)&sBt[buf][bc][2*bp] = p;
    }
  };

  // prologue: stage first tile
  issue_loads(0);
  store_lds(0);
  __syncthreads();

  for (int kb = 0; kb < K; kb += 32){
    const int buf  = (kb >> 5) & 1;
    const bool more = (kb + 32) < K;
    if (more) issue_loads(kb + 32);          // global loads overlap WMMA below

    // A fragment: two 16B chunks -> 2x ds_load_b128
    v8h alo = *(const v8h*)&sA[buf][am][akb];
    v8h ahi = *(const v8h*)&sA[buf][am][akb + 16];
    v16h afrag = __builtin_shufflevector(alo, ahi,
        0,1,2,3,4,5,6,7,8,9,10,11,12,13,14,15);
#pragma unroll
    for (int t = 0; t < 4; ++t){
      const int bn = t*16 + (lane & 15);
      const v8h* pb = (const v8h*)&sBt[buf][bn][bkb];
      v16h bfrag = __builtin_shufflevector(pb[0], pb[2],
          0,1,2,3,4,5,6,7,8,9,10,11,12,13,14,15);
      acc[t] = __builtin_amdgcn_wmma_f32_16x16x32_f16(false, afrag, false, bfrag,
                                                      (short)0, acc[t], false, false);
    }

    if (more) store_lds(buf ^ 1);            // fill the other buffer
    __syncthreads();                         // one barrier per K-step
  }

  // C/D layout: vgpr v -> M = v + 8*(l/16), N = l%16
#pragma unroll
  for (int t = 0; t < 4; ++t){
    const int cn = col0 + t*16 + (lane & 15);
    const float bv = bias[cn];
#pragma unroll
    for (int v = 0; v < 8; ++v){
      const int cm = row0 + wave*16 + v + ((lane >> 4) << 3);
      if (cm < M) C[(size_t)cm*Nn + cn] = acc[t][v] + bv;
    }
  }
}

// ---- edge attention logits: one wave per (edge, head) ----
// alpha[e,h] = (q[dst] . (k[src] + ea[e]@We[:,h])) / sqrt(DH); atomicMax segment max
template<int HEADS, int DH>
__global__ __launch_bounds__(256)
void edge_alpha_kernel(const float* __restrict__ q, const float* __restrict__ k,
                       const float* __restrict__ ea, const float* __restrict__ We,
                       const int* __restrict__ src, const int* __restrict__ dst,
                       float* __restrict__ alpha, unsigned* __restrict__ amaxk, int nE){
  constexpr int ROW = HEADS * DH;
  constexpr int PER = DH / 32;
  const int gwave = (blockIdx.x * blockDim.x + threadIdx.x) >> 5;
  const int lane  = threadIdx.x & 31;
  if (gwave >= nE * HEADS) return;               // wave-uniform exit
  const int e = gwave / HEADS;
  const int h = gwave % HEADS;
  const int s = src[e];
  const int d = dst[e];
  float earow[EDGE_DIM];
#pragma unroll
  for (int j = 0; j < EDGE_DIM; j += 4){
    float4 e4 = *(const float4*)&ea[(size_t)e*EDGE_DIM + j];
    earow[j] = e4.x; earow[j+1] = e4.y; earow[j+2] = e4.z; earow[j+3] = e4.w;
  }
  float acc = 0.f;
#pragma unroll
  for (int i = 0; i < PER; ++i){
    const int c = h*DH + lane + 32*i;
    float ev = 0.f;
#pragma unroll
    for (int j = 0; j < EDGE_DIM; ++j) ev += earow[j] * We[j*ROW + c];
    acc += q[(size_t)d*ROW + c] * (k[(size_t)s*ROW + c] + ev);
  }
#pragma unroll
  for (int o = 16; o >= 1; o >>= 1) acc += __shfl_xor(acc, o, 32);
  if (lane == 0){
    const float a = acc * rsqrtf((float)DH);
    alpha[gwave] = a;
    atomicMax(&amaxk[(size_t)d*HEADS + h], f2ord(a));
  }
}

// ---- exp + segment-sum of denominators; ex overwrites alpha in place ----
template<int HEADS>
__global__ __launch_bounds__(256)
void edge_exp_kernel(const int* __restrict__ dst, float* __restrict__ alpha,
                     const unsigned* __restrict__ amaxk, float* __restrict__ den, int nE){
  int i = blockIdx.x * blockDim.x + threadIdx.x;
  if (i >= nE * HEADS) return;
  const int e = i / HEADS;
  const int h = i % HEADS;
  const int d = dst[e];
  const float m  = ord2f(amaxk[(size_t)d*HEADS + h]);
  const float ex = expf(alpha[i] - m);
  alpha[i] = ex;
  atomicAdd(&den[(size_t)d*HEADS + h], ex);
}

// ---- weighted message scatter: out[dst] += attn * (v[src] + ea@We) ----
template<int HEADS, int DH>
__global__ __launch_bounds__(256)
void edge_agg_kernel(const float* __restrict__ v, const float* __restrict__ ea,
                     const float* __restrict__ We, const int* __restrict__ src,
                     const int* __restrict__ dst, const float* __restrict__ ex,
                     const float* __restrict__ den, float* __restrict__ out, int nE){
  constexpr int ROW = HEADS * DH;
  constexpr int PER = DH / 32;
  const int gwave = (blockIdx.x * blockDim.x + threadIdx.x) >> 5;
  const int lane  = threadIdx.x & 31;
  if (gwave >= nE * HEADS) return;
  const int e = gwave / HEADS;
  const int h = gwave % HEADS;
  const int s = src[e];
  const int d = dst[e];
  const float attn = ex[gwave] / (den[(size_t)d*HEADS + h] + 1e-16f);
  float earow[EDGE_DIM];
#pragma unroll
  for (int j = 0; j < EDGE_DIM; j += 4){
    float4 e4 = *(const float4*)&ea[(size_t)e*EDGE_DIM + j];
    earow[j] = e4.x; earow[j+1] = e4.y; earow[j+2] = e4.z; earow[j+3] = e4.w;
  }
#pragma unroll
  for (int i = 0; i < PER; ++i){
    const int c = h*DH + lane + 32*i;
    float ev = 0.f;
#pragma unroll
    for (int j = 0; j < EDGE_DIM; ++j) ev += earow[j] * We[j*ROW + c];
    atomicAdd(&out[(size_t)d*ROW + c], attn * (v[(size_t)s*ROW + c] + ev));
  }
}

extern "C" void kernel_launch(void* const* d_in, const int* in_sizes, int n_in,
                              void* d_out, int out_size, void* d_ws, size_t ws_size,
                              hipStream_t stream){
  (void)in_sizes; (void)n_in; (void)out_size; (void)ws_size;
  const float* x   = (const float*)d_in[0];
  const int*   ei  = (const int*)  d_in[1];
  const float* ea  = (const float*)d_in[2];
  const float* Wq1 = (const float*)d_in[3];
  const float* bq1 = (const float*)d_in[4];
  const float* Wk1 = (const float*)d_in[5];
  const float* bk1 = (const float*)d_in[6];
  const float* Wv1 = (const float*)d_in[7];
  const float* bv1 = (const float*)d_in[8];
  const float* We1 = (const float*)d_in[9];
  const float* Ws1 = (const float*)d_in[10];
  const float* bs1 = (const float*)d_in[11];
  const float* Wq2 = (const float*)d_in[12];
  const float* bq2 = (const float*)d_in[13];
  const float* Wk2 = (const float*)d_in[14];
  const float* bk2 = (const float*)d_in[15];
  const float* Wv2 = (const float*)d_in[16];
  const float* bv2 = (const float*)d_in[17];
  const float* We2 = (const float*)d_in[18];
  const float* Ws2 = (const float*)d_in[19];
  const float* bs2 = (const float*)d_in[20];
  const int* src = ei;
  const int* dst = ei + NEDGES;
  float* zout = (float*)d_out;

  // workspace carve-out (256B aligned)
  char* ws = (char*)d_ws;
  size_t off = 0;
  auto carve = [&](size_t bytes)->char*{
    char* p = ws + off;
    off = (off + bytes + 255) & ~(size_t)255;
    return p;
  };
  const int ROW1 = H1 * HID;  // 1024
  float*    q1     = (float*)   carve((size_t)NNODES*ROW1*4);
  float*    k1     = (float*)   carve((size_t)NNODES*ROW1*4);
  float*    v1     = (float*)   carve((size_t)NNODES*ROW1*4);
  float*    g1     = (float*)   carve((size_t)NNODES*ROW1*4);  // skip -> agg -> relu -> h
  float*    alpha1 = (float*)   carve((size_t)NEDGES*H1*4);
  unsigned* amax1  = (unsigned*)carve((size_t)NNODES*H1*4);
  float*    den1   = (float*)   carve((size_t)NNODES*H1*4);
  float*    q2     = (float*)   carve((size_t)NNODES*OUTD*4);
  float*    k2     = (float*)   carve((size_t)NNODES*OUTD*4);
  float*    v2     = (float*)   carve((size_t)NNODES*OUTD*4);
  float*    alpha2 = (float*)   carve((size_t)NEDGES*4);
  unsigned* amax2  = (unsigned*)carve((size_t)NNODES*4);
  float*    den2   = (float*)   carve((size_t)NNODES*4);

  const dim3 gblk(128);
  const dim3 grid1(ROW1/64, (NNODES + 63)/64);
  // ---- layer 1 projections (skip goes straight into agg buffer g1) ----
  gemm_bias_wmma_kernel<<<grid1, gblk, 0, stream>>>(x, Wq1, bq1, q1, NNODES, IN_DIM, ROW1);
  gemm_bias_wmma_kernel<<<grid1, gblk, 0, stream>>>(x, Wk1, bk1, k1, NNODES, IN_DIM, ROW1);
  gemm_bias_wmma_kernel<<<grid1, gblk, 0, stream>>>(x, Wv1, bv1, v1, NNODES, IN_DIM, ROW1);
  gemm_bias_wmma_kernel<<<grid1, gblk, 0, stream>>>(x, Ws1, bs1, g1, NNODES, IN_DIM, ROW1);

  // ---- layer 1 attention ----
  zero_u32_kernel<<<(NNODES*H1 + 255)/256, 256, 0, stream>>>(amax1, NNODES*H1);
  zero_u32_kernel<<<(NNODES*H1 + 255)/256, 256, 0, stream>>>((unsigned*)den1, NNODES*H1);
  const int tasks1 = NEDGES * H1;
  edge_alpha_kernel<H1, HID><<<(tasks1 + 7)/8, 256, 0, stream>>>(
      q1, k1, ea, We1, src, dst, alpha1, amax1, NEDGES);
  edge_exp_kernel<H1><<<(tasks1 + 255)/256, 256, 0, stream>>>(
      dst, alpha1, amax1, den1, NEDGES);
  edge_agg_kernel<H1, HID><<<(tasks1 + 7)/8, 256, 0, stream>>>(
      v1, ea, We1, src, dst, alpha1, den1, g1, NEDGES);
  relu_kernel<<<(NNODES*ROW1 + 255)/256, 256, 0, stream>>>(g1, NNODES*ROW1);

  // ---- layer 2 projections (skip goes straight into d_out) ----
  const dim3 grid2(OUTD/64, (NNODES + 63)/64);
  gemm_bias_wmma_kernel<<<grid2, gblk, 0, stream>>>(g1, Wq2, bq2, q2,   NNODES, ROW1, OUTD);
  gemm_bias_wmma_kernel<<<grid2, gblk, 0, stream>>>(g1, Wk2, bk2, k2,   NNODES, ROW1, OUTD);
  gemm_bias_wmma_kernel<<<grid2, gblk, 0, stream>>>(g1, Wv2, bv2, v2,   NNODES, ROW1, OUTD);
  gemm_bias_wmma_kernel<<<grid2, gblk, 0, stream>>>(g1, Ws2, bs2, zout, NNODES, ROW1, OUTD);

  // ---- layer 2 attention (H=1, mean over 1 head == identity) ----
  zero_u32_kernel<<<(NNODES + 255)/256, 256, 0, stream>>>(amax2, NNODES);
  zero_u32_kernel<<<(NNODES + 255)/256, 256, 0, stream>>>((unsigned*)den2, NNODES);
  edge_alpha_kernel<1, OUTD><<<(NEDGES + 7)/8, 256, 0, stream>>>(
      q2, k2, ea, We2, src, dst, alpha2, amax2, NEDGES);
  edge_exp_kernel<1><<<(NEDGES + 255)/256, 256, 0, stream>>>(
      dst, alpha2, amax2, den2, NEDGES);
  edge_agg_kernel<1, OUTD><<<(NEDGES + 7)/8, 256, 0, stream>>>(
      v2, ea, We2, src, dst, alpha2, den2, zout, NEDGES);
}
